// GRU_29102698397789
// MI455X (gfx1250) — matrix operand order repaired and checked by reading, not compiled, not run on hardware
//
#include <hip/hip_runtime.h>
#include <hip/hip_bf16.h>
#include <math.h>

#define BDIM 4096
#define NNODE 32
#define HDIM 512

#define GTM 64    // block tile rows
#define GTN 128   // block tile cols
#define GKS 32    // K step
#define APAD 40   // LDS row stride in bf16 (80B: 16B-aligned, bank-conflict-free)

typedef __attribute__((ext_vector_type(16))) __bf16 v16bf;
typedef __attribute__((ext_vector_type(8)))  float  v8f;

union Frag { v16bf v; uint4 q2[2]; };

__device__ __forceinline__ unsigned short f2bf(float f) {
  unsigned int u = __float_as_uint(f);
  u += 0x7FFFu + ((u >> 16) & 1u);   // round-to-nearest-even
  return (unsigned short)(u >> 16);
}
__device__ __forceinline__ float bf2f(unsigned short u) {
  return __uint_as_float((unsigned int)u << 16);
}

__device__ __forceinline__ void async_copy16(const unsigned short* g, const unsigned short* l) {
  // GLOBAL_LOAD_ASYNC_TO_LDS_B128: vdst = LDS byte address, vaddr = 64-bit global
  asm volatile("global_load_async_to_lds_b128 %0, %1, off"
               :: "v"((unsigned int)(uintptr_t)l),
                  "v"((unsigned long long)(uintptr_t)g)
               : "memory");
}
__device__ __forceinline__ void wait_async0() {
  asm volatile("s_wait_asynccnt 0x0" ::: "memory");
}

// ---------------------------------------------------------------------------
// C[M,N] = act( A[M,K] @ W[N,K]^T + bias );  A,W bf16, C f32 or bf16.
// Block = 128 threads (4 waves, 2x2) computing a 64x128 tile; A/B panels are
// double-buffered in LDS via async global->LDS copies; each wave runs
// 2x4 = 8 WMMAs per K-step from ds_load_b128 fragments.
//   rowidx: optional A row gather; crowmul: C row stride mult; act 1=leaky.
// grid = (M/64, N/128).
// ---------------------------------------------------------------------------
__global__ void __launch_bounds__(128) gemm_wmma_bf16(
    const unsigned short* __restrict__ A, int lda, const int* __restrict__ rowidx,
    const unsigned short* __restrict__ W, int ldw,
    const float* __restrict__ bias,
    void* __restrict__ Cv, int ldc, int crowmul,
    int K, int act, int outbf)
{
  __shared__ unsigned short As[2][GTM * APAD];   // 2 x 5120 B
  __shared__ unsigned short Bs[2][GTN * APAD];   // 2 x 10240 B

  const int tid  = threadIdx.x;
  const int lane = tid & 31;
  const int wave = tid >> 5;
  const int wr   = wave >> 1;   // row half (0/1)
  const int wc   = wave & 1;    // col half (0/1)
  const int hv   = lane >> 4;
  const int q    = lane & 15;

  const int m0 = blockIdx.x * GTM;
  const int n0 = blockIdx.y * GTN;

  // Per-thread source coordinates for stage copies:
  // A: 256 16B chunks -> 2/thread ; B: 512 16B chunks -> 4/thread
  const int ar0 = tid >> 2,            ak0 = (tid & 3) * 8;
  const int ar1 = (tid + 128) >> 2,    ak1 = ak0;
  const long arowA0 = (rowidx ? (long)rowidx[m0 + ar0] : (long)(m0 + ar0)) * (long)lda;
  const long arowA1 = (rowidx ? (long)rowidx[m0 + ar1] : (long)(m0 + ar1)) * (long)lda;

  const int nk = K / GKS;

  auto stage = [&](int buf, int k0) {
    async_copy16(A + arowA0 + k0 + ak0, &As[buf][ar0 * APAD + ak0]);
    async_copy16(A + arowA1 + k0 + ak1, &As[buf][ar1 * APAD + ak1]);
#pragma unroll
    for (int j = 0; j < 4; ++j) {
      const int u  = tid + 128 * j;
      const int c  = u >> 2;
      const int kc = (u & 3) * 8;
      async_copy16(W + (long)(n0 + c) * (long)ldw + k0 + kc, &Bs[buf][c * APAD + kc]);
    }
  };

  stage(0, 0);
  wait_async0();
  __syncthreads();

  v8f acc[2][4] = {{{}, {}, {}, {}}, {{}, {}, {}, {}}};

  for (int ks = 0; ks < nk; ++ks) {
    const int buf = ks & 1;
    if (ks + 1 < nk) stage(buf ^ 1, (ks + 1) * GKS);

    Frag fa[2];
#pragma unroll
    for (int rt = 0; rt < 2; ++rt) {
      // A 16x32: lane<16 -> K {0..7,16..23}; lane>=16 -> {8..15,24..31}
      const unsigned short* p = &As[buf][(wr * 32 + rt * 16 + q) * APAD + hv * 8];
      fa[rt].q2[0] = *(const uint4*)(p);
      fa[rt].q2[1] = *(const uint4*)(p + 16);
    }
#pragma unroll
    for (int t = 0; t < 4; ++t) {
      // B 32x16: lane<16 -> K {0..7,16..23} of col q; lane>=16 -> {8..15,24..31}
      const unsigned short* p = &Bs[buf][(wc * 64 + t * 16 + q) * APAD + hv * 8];
      Frag fb;
      fb.q2[0] = *(const uint4*)(p);
      fb.q2[1] = *(const uint4*)(p + 16);
      acc[0][t] = __builtin_amdgcn_wmma_f32_16x16x32_bf16(
          false, fa[0].v, false, fb.v, (short)0, acc[0][t], false, false);
      acc[1][t] = __builtin_amdgcn_wmma_f32_16x16x32_bf16(
          false, fa[1].v, false, fb.v, (short)0, acc[1][t], false, false);
    }
    if (ks + 1 < nk) wait_async0();
    __syncthreads();
  }

#pragma unroll
  for (int rt = 0; rt < 2; ++rt) {
#pragma unroll
    for (int t = 0; t < 4; ++t) {
      const int ncol = n0 + wc * 64 + t * 16 + q;
      const float bb = bias ? bias[ncol] : 0.0f;
#pragma unroll
      for (int j = 0; j < 8; ++j) {
        const int gm = m0 + wr * 32 + rt * 16 + j + hv * 8;  // C/D: VGPR j -> M=j (+8 upper)
        float vv = acc[rt][t][j] + bb;
        if (act == 1) vv = vv > 0.0f ? vv : 0.01f * vv;
        const long cidx = ((long)gm * crowmul) * (long)ldc + ncol;
        if (outbf) ((unsigned short*)Cv)[cidx] = f2bf(vv);
        else       ((float*)Cv)[cidx] = vv;
      }
    }
  }
}

// --------------------------- helper kernels --------------------------------
__global__ void zero_f32(float* p, long n) {
  long i = (long)blockIdx.x * blockDim.x + threadIdx.x;
  if (i < n) p[i] = 0.0f;
}
__global__ void zero_u32(unsigned* p, long n) {
  long i = (long)blockIdx.x * blockDim.x + threadIdx.x;
  if (i < n) p[i] = 0u;
}
__global__ void conv_f32_bf16(const float* __restrict__ s, unsigned short* __restrict__ d, long n) {
  long i = (long)blockIdx.x * blockDim.x + threadIdx.x;
  if (i < n) d[i] = f2bf(s[i]);
}

// mask_bits[b*32 + dst] |= 1<<src  (binary mask: .set semantics)
__global__ void edge_kernel(const int* __restrict__ ei, unsigned* __restrict__ mb) {
  int e = blockIdx.x * blockDim.x + threadIdx.x;
  if (e >= BDIM * 4) return;
  int e0 = ei[e];                 // batch*N + src
  int e1 = ei[BDIM * 4 + e];      // batch*N + dst
  atomicOr(&mb[(e0 / NNODE) * NNODE + (e1 % NNODE)], 1u << (e0 % NNODE));
}

// fidx[b] = b*32 + #{i : deg(b,i)>0}
__global__ void last_kernel(const unsigned* __restrict__ mb, int* __restrict__ fidx) {
  int b = blockIdx.x * blockDim.x + threadIdx.x;
  if (b >= BDIM) return;
  int cnt = 0;
#pragma unroll
  for (int i = 0; i < NNODE; ++i) cnt += (mb[b * NNODE + i] != 0u) ? 1 : 0;
  fidx[b] = b * NNODE + cnt;
}

// Stack gate weights (bf16): G[2048,1536] rows = [Wr|Ur],[Wz|Uz],[W|0],[0|U]
__global__ void buildG_kernel(
    const float* Wr_w, const float* Ur_w, const float* Wz_w, const float* Uz_w,
    const float* W_w,  const float* U_w,
    const float* Wr_b, const float* Ur_b, const float* Wz_b, const float* Uz_b,
    const float* W_b,  const float* U_b,
    unsigned short* __restrict__ G, float* __restrict__ Gb)
{
  long idx = (long)blockIdx.x * blockDim.x + threadIdx.x;
  if (idx >= (long)2048 * 1536) return;
  int r = (int)(idx / 1536);
  int c = (int)(idx % 1536);
  float v;
  if (r < 512)       v = (c < 1024) ? Wr_w[r * 1024 + c]            : Ur_w[r * 512 + (c - 1024)];
  else if (r < 1024) { int rr = r - 512;  v = (c < 1024) ? Wz_w[rr * 1024 + c] : Uz_w[rr * 512 + (c - 1024)]; }
  else if (r < 1536) { int rr = r - 1024; v = (c < 1024) ? W_w[rr * 1024 + c]  : 0.0f; }
  else               { int rr = r - 1536; v = (c < 1024) ? 0.0f                : U_w[rr * 512 + (c - 1024)]; }
  G[idx] = f2bf(v);
  if (c == 0) {
    float bb;
    if (r < 512)       bb = Wr_b[r] + Ur_b[r];
    else if (r < 1024) bb = Wz_b[r - 512] + Uz_b[r - 512];
    else if (r < 1536) bb = W_b[r - 1024];
    else               bb = U_b[r - 1536];
    Gb[r] = bb;
  }
}

// In-place LayerNorm (f32 statistics) on bf16 node[B*N, 512]
__global__ void __launch_bounds__(128) ln_kernel(
    unsigned short* __restrict__ node, const float* __restrict__ g, const float* __restrict__ b)
{
  __shared__ float red[128];
  unsigned short* p = node + (long)blockIdx.x * HDIM;
  float s = 0.0f, ss = 0.0f;
  for (int h = threadIdx.x; h < HDIM; h += 128) { float x = bf2f(p[h]); s += x; ss += x * x; }
  red[threadIdx.x] = s; __syncthreads();
  for (int o = 64; o > 0; o >>= 1) { if (threadIdx.x < o) red[threadIdx.x] += red[threadIdx.x + o]; __syncthreads(); }
  float mean = red[0] / (float)HDIM; __syncthreads();
  red[threadIdx.x] = ss; __syncthreads();
  for (int o = 64; o > 0; o >>= 1) { if (threadIdx.x < o) red[threadIdx.x] += red[threadIdx.x + o]; __syncthreads(); }
  float var  = red[0] / (float)HDIM - mean * mean;
  float rstd = rsqrtf(var + 1e-6f);
  for (int h = threadIdx.x; h < HDIM; h += 128)
    p[h] = f2bf((bf2f(p[h]) - mean) * rstd * g[h] + b[h]);
}

// Per step i: h_i = bitmask gather-sum of states; comb=[x_i|h_i|h_i] in bf16
__global__ void hi_comb_kernel(
    const float* __restrict__ states, const unsigned short* __restrict__ node,
    const unsigned* __restrict__ mb,
    float* __restrict__ h_i, unsigned short* __restrict__ comb, int i)
{
  int b = blockIdx.x;
  unsigned m = mb[b * NNODE + i];
  const float* srow = states + (long)b * NNODE * HDIM;
  const unsigned short* xrow = node + ((long)b * NNODE + i) * HDIM;
  unsigned short* crow = comb + (long)b * 1536;
  for (int h = threadIdx.x; h < HDIM; h += 256) {
    float s = 0.0f;
    unsigned mm = m;
    while (mm) { int si = __ffs(mm) - 1; mm &= mm - 1; s += srow[si * HDIM + h]; }
    unsigned short sb = f2bf(s);
    crow[h]        = xrow[h];
    crow[512 + h]  = sb;
    crow[1024 + h] = sb;
    h_i[(long)b * HDIM + h] = s;
  }
}

// GRU update; gates[b,0:512]=r_pre,[512:1024]=z_pre,[1024:1536]=cW,[1536:2048]=cU
__global__ void gru_update_kernel(
    const float* __restrict__ gates, const float* __restrict__ h_i,
    float* __restrict__ states, int i)
{
  long idx = (long)blockIdx.x * blockDim.x + threadIdx.x;
  if (idx >= (long)BDIM * HDIM) return;
  int b  = (int)(idx >> 9);
  int hh = (int)(idx & 511);
  const float* grow = gates + (long)b * 2048;
  float r  = 1.0f / (1.0f + expf(-grow[hh]));
  float z  = 1.0f / (1.0f + expf(-grow[512 + hh]));
  float ht = tanhf(grow[1024 + hh] + r * grow[1536 + hh]);
  float hi = h_i[(long)b * HDIM + hh];
  states[((long)b * NNODE + i) * HDIM + hh] = (1.0f - z) * hi + z * ht;
}

// fin_bf[b,:] = bf16(states[fidx[b], :])
__global__ void gather_conv_kernel(
    const float* __restrict__ states, const int* __restrict__ fidx,
    unsigned short* __restrict__ out)
{
  long i = (long)blockIdx.x * blockDim.x + threadIdx.x;
  if (i >= (long)BDIM * HDIM) return;
  int b  = (int)(i >> 9);
  int hh = (int)(i & 511);
  out[i] = f2bf(states[(long)fidx[b] * HDIM + hh]);
}

// logits[b] = Wo2_b + dot(hid[b,:], Wo2_w)
__global__ void logits_kernel(
    const float* __restrict__ hid, const float* __restrict__ w2,
    const float* __restrict__ b2, float* __restrict__ out)
{
  int b = blockIdx.x * blockDim.x + threadIdx.x;
  if (b >= BDIM) return;
  const float* p = hid + (long)b * HDIM;
  float s = b2[0];
  for (int h = 0; h < HDIM; ++h) s += p[h] * w2[h];
  out[b] = s;
}

// ---------------------------------------------------------------------------
extern "C" void kernel_launch(void* const* d_in, const int* in_sizes, int n_in,
                              void* d_out, int out_size, void* d_ws, size_t ws_size,
                              hipStream_t stream)
{
  (void)in_sizes; (void)n_in; (void)out_size; (void)ws_size;
  const float* t      = (const float*)d_in[0];
  const float* v      = (const float*)d_in[1];
  const int*   m_idx  = (const int*)d_in[2];
  const int*   edge   = (const int*)d_in[3];
  const float* Wr_w   = (const float*)d_in[5];
  const float* Wr_b   = (const float*)d_in[6];
  const float* Ur_w   = (const float*)d_in[7];
  const float* Ur_b   = (const float*)d_in[8];
  const float* Wz_w   = (const float*)d_in[9];
  const float* Wz_b   = (const float*)d_in[10];
  const float* Uz_w   = (const float*)d_in[11];
  const float* Uz_b   = (const float*)d_in[12];
  const float* W_w    = (const float*)d_in[13];
  const float* W_b    = (const float*)d_in[14];
  const float* U_w    = (const float*)d_in[15];
  const float* U_b    = (const float*)d_in[16];
  const float* Wo1_w  = (const float*)d_in[17];
  const float* Wo1_b  = (const float*)d_in[18];
  const float* Wo2_w  = (const float*)d_in[19];
  const float* Wo2_b  = (const float*)d_in[20];
  const float* Wst_w  = (const float*)d_in[21];
  const float* Wst_b  = (const float*)d_in[22];
  const float* emb    = (const float*)d_in[23];
  const float* tp_w   = (const float*)d_in[24];
  const float* tp_b   = (const float*)d_in[25];
  const float* vp_w   = (const float*)d_in[26];
  const float* vp_b   = (const float*)d_in[27];
  const float* np_w   = (const float*)d_in[28];
  const float* np_b   = (const float*)d_in[29];
  const float* ln_g   = (const float*)d_in[30];
  const float* ln_b   = (const float*)d_in[31];

  // ---- workspace layout ----
  char* base = (char*)d_ws;
  size_t off = 0;
  auto allocF = [&](size_t n) { float* p = (float*)(base + off); off += n * 4; return p; };
  auto allocH = [&](size_t n) { unsigned short* p = (unsigned short*)(base + off); off += n * 2; return p; };

  float* states = allocF((size_t)BDIM * NNODE * HDIM);   // 256 MB
  float* gates  = allocF((size_t)BDIM * 2048);
  float* h_i    = allocF((size_t)BDIM * HDIM);
  float* hid    = allocF((size_t)BDIM * HDIM);
  float* Gb     = allocF(2048);
  unsigned short* node_bf = allocH((size_t)BDIM * NNODE * HDIM);  // 128 MB
  unsigned short* comb_bf = allocH((size_t)BDIM * 1536);
  unsigned short* G_bf    = allocH((size_t)2048 * 1536);
  unsigned short* vete_bf = allocH((size_t)BDIM * 1024);
  unsigned short* t_bf    = allocH((size_t)BDIM * 768);
  unsigned short* v_bf    = allocH((size_t)BDIM * 768);
  unsigned short* emb_bf  = allocH((size_t)201 * 512);
  unsigned short* vpw_bf  = allocH((size_t)512 * 768);
  unsigned short* tpw_bf  = allocH((size_t)512 * 768);
  unsigned short* npw_bf  = allocH((size_t)512 * 512);
  unsigned short* Wst_bf  = allocH((size_t)512 * 1024);
  unsigned short* Wo1_bf  = allocH((size_t)512 * 512);
  unsigned short* fin_bf  = allocH((size_t)BDIM * HDIM);
  unsigned* mb = (unsigned*)(base + off); off += (size_t)BDIM * NNODE * 4;
  int* fidx    = (int*)(base + off);      off += (size_t)BDIM * 4;

  auto cgrid = [](long n) { return (unsigned)((n + 255) / 256); };

  // ---- prep ----
  const long nStates = (long)BDIM * NNODE * HDIM;
  zero_f32<<<cgrid(nStates), 256, 0, stream>>>(states, nStates);
  zero_u32<<<cgrid(BDIM * NNODE), 256, 0, stream>>>(mb, (long)BDIM * NNODE);
  edge_kernel<<<cgrid(BDIM * 4), 256, 0, stream>>>(edge, mb);
  last_kernel<<<cgrid(BDIM), 256, 0, stream>>>(mb, fidx);
  buildG_kernel<<<cgrid((long)2048 * 1536), 256, 0, stream>>>(
      Wr_w, Ur_w, Wz_w, Uz_w, W_w, U_w, Wr_b, Ur_b, Wz_b, Uz_b, W_b, U_b, G_bf, Gb);
  conv_f32_bf16<<<cgrid((long)BDIM * 768), 256, 0, stream>>>(t, t_bf, (long)BDIM * 768);
  conv_f32_bf16<<<cgrid((long)BDIM * 768), 256, 0, stream>>>(v, v_bf, (long)BDIM * 768);
  conv_f32_bf16<<<cgrid((long)201 * 512), 256, 0, stream>>>(emb, emb_bf, (long)201 * 512);
  conv_f32_bf16<<<cgrid((long)512 * 768), 256, 0, stream>>>(vp_w, vpw_bf, (long)512 * 768);
  conv_f32_bf16<<<cgrid((long)512 * 768), 256, 0, stream>>>(tp_w, tpw_bf, (long)512 * 768);
  conv_f32_bf16<<<cgrid((long)512 * 512), 256, 0, stream>>>(np_w, npw_bf, (long)512 * 512);
  conv_f32_bf16<<<cgrid((long)512 * 1024), 256, 0, stream>>>(Wst_w, Wst_bf, (long)512 * 1024);
  conv_f32_bf16<<<cgrid((long)512 * 512), 256, 0, stream>>>(Wo1_w, Wo1_bf, (long)512 * 512);

  // ---- projections (all WMMA) ----
  // v_e -> vete[:,0:512], t_e -> vete[:,512:1024]   (concat([v_e, t_e]))
  gemm_wmma_bf16<<<dim3(BDIM / GTM, 512 / GTN), 128, 0, stream>>>(
      v_bf, 768, nullptr, vpw_bf, 768, vp_b, vete_bf, 1024, 1, 768, 0, 1);
  gemm_wmma_bf16<<<dim3(BDIM / GTM, 512 / GTN), 128, 0, stream>>>(
      t_bf, 768, nullptr, tpw_bf, 768, tp_b, vete_bf + 512, 1024, 1, 768, 0, 1);
  // node = nproj(emb[m_idx])  (row gather)
  gemm_wmma_bf16<<<dim3((BDIM * NNODE) / GTM, 512 / GTN), 128, 0, stream>>>(
      emb_bf, 512, m_idx, npw_bf, 512, np_b, node_bf, 512, 1, 512, 0, 1);
  // node[:,0,:] = Wstart([v_e|t_e])  (C row stride 32)
  gemm_wmma_bf16<<<dim3(BDIM / GTM, 512 / GTN), 128, 0, stream>>>(
      vete_bf, 1024, nullptr, Wst_bf, 1024, Wst_b, node_bf, 512, NNODE, 1024, 0, 1);
  ln_kernel<<<BDIM * NNODE, 128, 0, stream>>>(node_bf, ln_g, ln_b);

  // ---- sequential GRU scan over N=32 nodes ----
  for (int i = 0; i < NNODE; ++i) {
    hi_comb_kernel<<<BDIM, 256, 0, stream>>>(states, node_bf, mb, h_i, comb_bf, i);
    gemm_wmma_bf16<<<dim3(BDIM / GTM, 2048 / GTN), 128, 0, stream>>>(
        comb_bf, 1536, nullptr, G_bf, 1536, Gb, gates, 2048, 1, 1536, 0, 0);
    gru_update_kernel<<<cgrid((long)BDIM * HDIM), 256, 0, stream>>>(gates, h_i, states, i);
  }

  // ---- head ----
  gather_conv_kernel<<<cgrid((long)BDIM * HDIM), 256, 0, stream>>>(states, fidx, fin_bf);
  gemm_wmma_bf16<<<dim3(BDIM / GTM, 512 / GTN), 128, 0, stream>>>(
      fin_bf, 512, nullptr, Wo1_bf, 512, Wo1_b, hid, 512, 1, 512, 1, 0);
  logits_kernel<<<cgrid(BDIM), 256, 0, stream>>>(hid, Wo2_w, Wo2_b, (float*)d_out);
}
